// Lorentz_fully_connected_mlr_angle_55490977464542
// MI455X (gfx1250) — matrix operand order repaired and checked by reading, not compiled, not run on hardware
//
#include <hip/hip_runtime.h>

typedef __bf16 v16bf __attribute__((ext_vector_type(16)));
typedef float  v8f   __attribute__((ext_vector_type(8)));
typedef int    v4i   __attribute__((ext_vector_type(4)));

#define AS1 __attribute__((address_space(1)))
#define AS3 __attribute__((address_space(3)))

constexpr int   IN      = 256;
constexpr int   OUT     = 64;
constexpr int   TILE_M  = 16;            // rows per wave tile
constexpr int   WAVES   = 8;             // waves per workgroup (256 threads, wave32)
constexpr int   WG_ROWS = WAVES * TILE_M;// 128 rows per workgroup macro-tile
constexpr int   KH      = 64;            // K-quarter staged in LDS per wave
constexpr int   NSTAGE  = IN / KH;       // 4
constexpr float SK   = 0.31622776601683794f;  // sqrt(0.1)
constexpr float RSK  = 3.16227766016837933f;  // 1/sqrt(0.1)
constexpr float FEPS = 1e-7f;

#if __has_builtin(__builtin_amdgcn_global_load_async_to_lds_b128)
#define HAVE_ASYNC_LDS 1
#else
#define HAVE_ASYNC_LDS 0
#endif

#if HAVE_ASYNC_LDS
// per-lane async copy: 16B global -> 16B LDS, tracked with ASYNCcnt
#define ASYNC_CP16(gsrc, ldst)                                               \
  __builtin_amdgcn_global_load_async_to_lds_b128(                            \
      (AS1 v4i*)((AS1 void*)(void*)(gsrc)),                                  \
      (AS3 v4i*)((AS3 void*)(void*)(ldst)), 0, 0)
#if __has_builtin(__builtin_amdgcn_s_wait_asynccnt)
#define WAIT_ASYNC(n) __builtin_amdgcn_s_wait_asynccnt(n)
#else
#define WAIT_ASYNC(n) asm volatile("s_wait_asynccnt %0" ::"i"(n) : "memory")
#endif
#define WAIT_DS0() asm volatile("s_wait_dscnt 0" ::: "memory")
#endif

__global__ __launch_bounds__(256) void lorentz_mlr_wmma(
    const float* __restrict__ x, const float* __restrict__ z,
    const float* __restrict__ a, float* __restrict__ out,
    int R, int numMacro)
{
  extern __shared__ __align__(16) char smem[];
  __bf16* zl  = (__bf16*)smem;                       // [OUT][IN] bf16  (32 KB)
  float*  znl = (float*)(smem + OUT * IN * 2);       // [OUT]
  float*  cal = znl + OUT;
  float*  t2l = cal + OUT;                           // sinh(sk*a)*||z||
  float*  xs_all = (float*)(smem + OUT * IN * 2 + 3 * OUT * 4); // [WAVES][2][TILE_M*KH]

  const int tid  = threadIdx.x;
  const int lane = tid & 31;
  const int wave = tid >> 5;
  const int hi   = lane >> 4;      // 0 or 1 (lane half)
  const int lo   = lane & 15;

  // ---------------- one-time per-WG setup ----------------
  #pragma unroll
  for (int i = 0; i < (OUT * IN / 4) / 256; ++i) {   // z -> bf16 LDS, coalesced
    int f4 = i * 256 + tid;
    float4 v = ((const float4*)z)[f4];
    union { __bf16 h[4]; uint2 u; } pk;
    pk.h[0] = (__bf16)v.x; pk.h[1] = (__bf16)v.y;
    pk.h[2] = (__bf16)v.z; pk.h[3] = (__bf16)v.w;
    *(uint2*)&zl[f4 * 4] = pk.u;
  }
  if (tid < OUT) {                                   // per-output constants
    float s = 0.f;
    const float4* zr = (const float4*)(z + tid * IN);
    #pragma unroll 8
    for (int k = 0; k < IN / 4; ++k) {
      float4 v = zr[k];
      s = fmaf(v.x, v.x, fmaf(v.y, v.y, fmaf(v.z, v.z, fmaf(v.w, v.w, s))));
    }
    float zn = fmaxf(sqrtf(s), FEPS);
    float t  = SK * a[tid];
    float e  = __expf(t), ei = __expf(-t);
    znl[tid] = zn;
    cal[tid] = 0.5f * (e + ei);
    t2l[tid] = 0.5f * (e - ei) * zn;
  }
  __syncthreads();

  float ca_t[4], t2_t[4], g_t[4], ig_t[4];
  #pragma unroll
  for (int t = 0; t < 4; ++t) {
    int o = t * 16 + lo;
    float zn = znl[o];
    ca_t[t] = cal[o];
    t2_t[t] = t2l[o];
    g_t[t]  = zn * RSK;   // zn/sk
    ig_t[t] = SK / zn;    // sk/zn
  }

  float* xs0 = xs_all + wave * (2 * TILE_M * KH);    // double-buffered stage
  float* xs1 = xs0 + TILE_M * KH;

  // ---------------- grid-stride over 128-row macro tiles ----------------
  for (int mt = blockIdx.x; mt < numMacro; mt += gridDim.x) {
    const int rowBase = mt * WG_ROWS + wave * TILE_M;
    const float* xg = x + (long long)rowBase * IN;

    if (mt + (int)gridDim.x < numMacro)   // warm L2 for the next macro tile
      __builtin_prefetch(x + ((long long)(mt + gridDim.x) * WG_ROWS +
                              wave * TILE_M + lo) * IN, 0, 1);

    v8f acc[4] = {};
    float sumsq = 0.f;

    // stage issue: 16 rows x KH f32, coalesced 2x256B per instruction
    auto stage_issue = [&](int s, float* buf) {
      #pragma unroll
      for (int i = 0; i < 8; ++i) {
        int row = i * 2 + hi;
        const float* gp = xg + row * IN + s * KH + lo * 4;
        float*       lp = buf + row * KH + lo * 4;
#if HAVE_ASYNC_LDS
        ASYNC_CP16(gp, lp);
#else
        *(float4*)lp = *(const float4*)gp;
#endif
      }
    };

    stage_issue(0, xs0);

    #pragma unroll
    for (int s = 0; s < NSTAGE; ++s) {
      float* cur = (s & 1) ? xs1 : xs0;
      if (s + 1 < NSTAGE) {
#if HAVE_ASYNC_LDS
        WAIT_DS0();   // prior ds_loads done before async writes reuse buffers
#endif
        stage_issue(s + 1, (s & 1) ? xs0 : xs1);
      }
#if HAVE_ASYNC_LDS
      if (s + 1 < NSTAGE) WAIT_ASYNC(8); else WAIT_ASYNC(0);
#endif
      #pragma unroll
      for (int kk = 0; kk < KH; kk += 32) {
        // A fragment, contiguous K order (matches B):
        //   lanes 0-15:  elems 0-15 = K+0..15
        //   lanes 16-31: elems 0-15 = K+16..31
        const float* ar = &cur[lo * KH + kk + hi * 16];
        float4 a0 = *(const float4*)(ar);
        float4 a1 = *(const float4*)(ar + 4);
        float4 a2 = *(const float4*)(ar + 8);
        float4 a3 = *(const float4*)(ar + 12);
        float av[16] = { a0.x, a0.y, a0.z, a0.w,  a1.x, a1.y, a1.z, a1.w,
                         a2.x, a2.y, a2.z, a2.w,  a3.x, a3.y, a3.z, a3.w };
        v16bf af;
        #pragma unroll
        for (int i = 0; i < 16; ++i) {
          sumsq = fmaf(av[i], av[i], sumsq);   // exact f32 row-norm accumulation
          af[i] = (__bf16)av[i];
        }
        const int kg = s * KH + kk + hi * 16;  // B: lane=col o, 16 consecutive K
        #pragma unroll
        for (int t = 0; t < 4; ++t) {
          int o = t * 16 + lo;
          v16bf bf = *(const v16bf*)&zl[o * IN + kg];
          acc[t] = __builtin_amdgcn_wmma_f32_16x16x32_bf16(
                       false, af, false, bf, (short)0, acc[t], false, false);
        }
      }
    }

    // ---------------- epilogue ----------------
    float full = sumsq + __shfl_xor(sumsq, 16);    // lanes l, l^16 hold K-halves
    float* orow = out + (long long)rowBase * OUT;

    #pragma unroll
    for (int j = 0; j < 8; ++j) {
      const int M = j + hi * 8;                 // row within tile (C layout)
      float rn2 = __shfl(full, M);              // row norm^2 lives in lane M
      float xn  = fmaxf(sqrtf(rn2), FEPS);
      float tt  = xn * RSK;                     // xn/sk
      float e   = __expf(tt), ei = __expf(-tt);
      float p_time = SK * 0.5f * (e + ei);      // sk*cosh(xn/sk)
      float scale  = SK * 0.5f * (e - ei) / xn; // sk*sinh(xn/sk)/xn
      #pragma unroll
      for (int t = 0; t < 4; ++t) {
        float dot   = acc[t][j];
        float inner = fmaf(dot * scale, ca_t[t], -p_time * t2_t[t]);
        float u  = inner * ig_t[t];
        float au = fabsf(u);
        float r = (au > 1e18f) ? (__logf(au) + 0.69314718f)
                               : __logf(au + sqrtf(fmaf(au, au, 1.f)));
        // streaming output: non-temporal, don't pollute L2
        __builtin_nontemporal_store(g_t[t] * copysignf(r, u),
                                    &orow[M * OUT + t * 16 + lo]);
      }
    }
  }
}

extern "C" void kernel_launch(void* const* d_in, const int* in_sizes, int n_in,
                              void* d_out, int out_size, void* d_ws, size_t ws_size,
                              hipStream_t stream) {
  const float* x = (const float*)d_in[0];
  const float* z = (const float*)d_in[1];
  const float* a = (const float*)d_in[2];
  float* out = (float*)d_out;

  const int R = in_sizes[0] / IN;           // B*S rows
  const int numMacro = R / WG_ROWS;         // 8192 for the reference shapes
  int grid = numMacro < 1024 ? (numMacro > 0 ? numMacro : 1) : 1024;

  const size_t shmem = (size_t)OUT * IN * 2                        // z bf16
                     + 3 * OUT * sizeof(float)                     // constants
                     + (size_t)WAVES * 2 * TILE_M * KH * sizeof(float); // x stage x2

  hipLaunchKernelGGL(lorentz_mlr_wmma, dim3(grid), dim3(256), shmem, stream,
                     x, z, a, out, R, numMacro);
}
